// SelfAttention_41884521070902
// MI455X (gfx1250) — compile-verified, matrix operand order checked by reading, not run on hardware
//
#include <hip/hip_runtime.h>

// Problem constants (from reference)
#define BB 4
#define SS 2048
#define EE 1024
#define HH 16
#define DD 64
#define MROWS (BB * SS)      // 8192
#define N_QKV (3 * EE)       // 3072

typedef __attribute__((ext_vector_type(16))) __bf16 v16bf;
typedef __attribute__((ext_vector_type(8)))  __bf16 bf16x8;
typedef __attribute__((ext_vector_type(8)))  float  v8f;

// ---------------------------------------------------------------------------
// DPP16-based 16-lane butterfly reduction (pure VALU).
//   0xB1 = quad_perm(1,0,3,2), 0x4E = quad_perm(2,3,0,1),
//   0x141 = row_half_mirror,   0x140 = row_mirror
// ---------------------------------------------------------------------------
template <int CTRL>
__device__ __forceinline__ float dppf(float v) {
    return __int_as_float(__builtin_amdgcn_update_dpp(
        0, __float_as_int(v), CTRL, 0xF, 0xF, true));
}
__device__ __forceinline__ float rowmax16(float v) {
    v = fmaxf(v, dppf<0xB1>(v));
    v = fmaxf(v, dppf<0x4E>(v));
    v = fmaxf(v, dppf<0x141>(v));
    v = fmaxf(v, dppf<0x140>(v));
    return v;
}
__device__ __forceinline__ float rowsum16(float v) {
    v += dppf<0xB1>(v);
    v += dppf<0x4E>(v);
    v += dppf<0x141>(v);
    v += dppf<0x140>(v);
    return v;
}

// ---------------------------------------------------------------------------
// Async global -> LDS copy, 16 bytes per lane (GLOBAL_LOAD_ASYNC_TO_LDS_B128,
// tracked by ASYNCcnt). Builtin signature (from compiler diagnostic): first
// param is int4* in the global (AS1) address space; LDS side is AS3.
// Generic LDS pointer low 32 bits == LDS byte offset.
// ---------------------------------------------------------------------------
typedef __attribute__((__vector_size__(16))) int vi4;
typedef __attribute__((address_space(1))) vi4* gvi4_p;
typedef __attribute__((address_space(3))) vi4* svi4_p;

__device__ __forceinline__ void async_b128(const __bf16* g, __bf16* l) {
    unsigned lds_off = (unsigned)(unsigned long long)l;
#if __has_builtin(__builtin_amdgcn_global_load_async_to_lds_b128)
    __builtin_amdgcn_global_load_async_to_lds_b128(
        (gvi4_p)(unsigned long long)g,
        (svi4_p)(unsigned long long)lds_off, 0, 0);
#else
    asm volatile("global_load_async_to_lds_b128 %0, %1, off"
                 :: "v"(lds_off), "v"(g) : "memory");
#endif
}

// ---------------------------------------------------------------------------
// Fragment loader for 16-bit A/B matrices (CDNA5 WMMA 16x16x32 layout):
// lane L holds row (L&15); v16bf elements 0..7 are K = k0 + half*8 .. +7,
// elements 8..15 are K = k0 + 16 + half*8 .. +7   (half = L>>4).
// Works for global or LDS (generic) row-major [row][K] bf16, stride `ld`.
// ---------------------------------------------------------------------------
__device__ __forceinline__ v16bf load_frag(const __bf16* base,
                                           int row0, int ld, int k0, int lane) {
    int r    = row0 + (lane & 15);
    int half = (lane >> 4) & 1;
    const __bf16* p = base + (size_t)r * ld + k0 + half * 8;
    bf16x8 lo = *reinterpret_cast<const bf16x8*>(p);
    bf16x8 hi = *reinterpret_cast<const bf16x8*>(p + 16);
    v16bf f;
#pragma unroll
    for (int i = 0; i < 8; ++i) { f[i] = lo[i]; f[8 + i] = hi[i]; }
    return f;
}

// ---------------------------------------------------------------------------
// f32 -> bf16 bulk convert (4 elems/thread)
// ---------------------------------------------------------------------------
__global__ void cvt_f32_bf16(const float* __restrict__ src,
                             __bf16* __restrict__ dst, int n) {
    int i = (blockIdx.x * blockDim.x + threadIdx.x) * 4;
    if (i + 3 < n) {
        float4 v = *reinterpret_cast<const float4*>(src + i);
        dst[i + 0] = (__bf16)v.x;
        dst[i + 1] = (__bf16)v.y;
        dst[i + 2] = (__bf16)v.z;
        dst[i + 3] = (__bf16)v.w;
    }
}

// src is [rows(K) x cols(N)] f32 row-major; dst is [cols(N) x rows(K)] bf16 (B^T)
__global__ void transpose_cvt(const float* __restrict__ src,
                              __bf16* __restrict__ dst, int rows, int cols) {
    int idx = blockIdx.x * blockDim.x + threadIdx.x;
    if (idx >= rows * cols) return;
    int k = idx % rows;
    int n = idx / rows;
    dst[(size_t)n * rows + k] = (__bf16)src[(size_t)k * cols + n];
}

// ---------------------------------------------------------------------------
// WMMA GEMM: C[M,N] = A[M,K]_bf16 @ Bt[N,K]_bf16^T
// One wave computes a 16(M) x 64(N) tile (4 C-fragments), K-loop step 32.
// MODE 0: QKV epilogue -> scatter q,k to [B,H,S,D], v to [B,H,D,S] (all bf16)
// MODE 1: out epilogue -> f32 C + bias, row-major [M,N]
// ---------------------------------------------------------------------------
template <int MODE>
__global__ void gemm_wave(const __bf16* __restrict__ A,
                          const __bf16* __restrict__ Bt,
                          int N, int K,
                          __bf16* __restrict__ qb, __bf16* __restrict__ kb,
                          __bf16* __restrict__ vt,
                          const float* __restrict__ bias,
                          float* __restrict__ Cout) {
    int lane  = threadIdx.x & 31;
    int wave  = threadIdx.x >> 5;
    int m0    = blockIdx.x * 16;
    int ntile = blockIdx.y * 8 + wave;   // 64 cols per wave
    int n0    = ntile * 64;
    if (n0 >= N) return;                 // wave-uniform

    v8f acc[4] = {};
    for (int k0 = 0; k0 < K; k0 += 32) {
        v16bf a = load_frag(A, m0, K, k0, lane);
        if (k0 + 32 < K)
            __builtin_prefetch(A + (size_t)m0 * K + k0 + 32, 0, 0);
#pragma unroll
        for (int j = 0; j < 4; ++j) {
            v16bf b = load_frag(Bt, n0 + j * 16, K, k0, lane);
            acc[j] = __builtin_amdgcn_wmma_f32_16x16x32_bf16(
                false, a, false, b, (short)0, acc[j], false, false);
        }
    }

    // Epilogue. C-frag: element i of lane L is row m0+(L>>4)*8+i, col n0+j*16+(L&15)
    int rowh = ((lane >> 4) & 1) * 8;
    int col  = lane & 15;
#pragma unroll
    for (int j = 0; j < 4; ++j) {
#pragma unroll
        for (int i = 0; i < 8; ++i) {
            int m = m0 + rowh + i;                 // global row in [0, B*S)
            int n = n0 + j * 16 + col;
            if (MODE == 0) {
                int sec = n >> 10;                 // 0:q 1:k 2:v
                int e   = n & (EE - 1);
                int h   = e >> 6;
                int d   = e & (DD - 1);
                int b   = m >> 11;                 // / S
                int s   = m & (SS - 1);
                __bf16 val = (__bf16)acc[j][i];
                if (sec == 0)
                    qb[(((size_t)(b * HH + h)) * SS + s) * DD + d] = val;
                else if (sec == 1)
                    kb[(((size_t)(b * HH + h)) * SS + s) * DD + d] = val;
                else
                    vt[(((size_t)(b * HH + h)) * DD + d) * SS + s] = val;
            } else {
                Cout[(size_t)m * N + n] = acc[j][i] + bias[n];
            }
        }
    }
}

// ---------------------------------------------------------------------------
// Flash attention, 4 waves (128 threads) per block.
// Block handles one (b,h) and 64 consecutive query rows (wave w -> 16 rows).
// K/V chunks (32 keys) are staged in LDS via async global->LDS DMA, double
// buffered, and shared by all 4 waves (4x less L2 read traffic). Online
// softmax in f32 with DPP reductions; P tile goes C-layout -> A-layout via a
// small per-wave LDS staging buffer.
// q,k: [B,H,S,D] bf16 ; v: [B,H,D,S] bf16 (transposed) ; out o: [B*S,E] bf16
// ---------------------------------------------------------------------------
__global__ void attn_kernel(const __bf16* __restrict__ qb,
                            const __bf16* __restrict__ kb,
                            const __bf16* __restrict__ vt,
                            __bf16* __restrict__ ob) {
    __shared__ __attribute__((aligned(16))) __bf16 kbuf[2][32 * DD]; // 2 x 4KB
    __shared__ __attribute__((aligned(16))) __bf16 vbuf[2][DD * 32]; // 2 x 4KB
    __shared__ __attribute__((aligned(16))) __bf16 p_lds[4][16 * 32];// 4 x 1KB

    int t    = threadIdx.x;              // 0..127
    int lane = t & 31;
    int wave = t >> 5;
    int qc   = blockIdx.x & (SS / 64 - 1);   // 32 query-chunks per head
    int bh   = blockIdx.x >> 5;              // b*H + h
    int q0   = qc * 64 + wave * 16;

    const __bf16* qbase = qb + (size_t)bh * SS * DD;
    const __bf16* kbase = kb + (size_t)bh * SS * DD;
    const __bf16* vbase = vt + (size_t)bh * DD * SS;

    // Q fragments (persist): contraction D=64 -> two K=32 frags
    v16bf qa0 = load_frag(qbase, q0, DD, 0, lane);
    v16bf qa1 = load_frag(qbase, q0, DD, 32, lane);

    float m[8], l[8];
    v8f   o[4] = {};
#pragma unroll
    for (int i = 0; i < 8; ++i) { m[i] = -1e30f; l[i] = 0.0f; }

    const float scale = 0.125f;          // D^-0.5
    int rowh = ((lane >> 4) & 1) * 8;
    int col  = lane & 15;
    __bf16* pw = &p_lds[wave][0];

    // Stage a 32-key chunk into LDS buffer `buf` (4 async b128 per thread:
    // K is contiguous 4KB; V^T rows are 32 elements = 64B each).
    auto stage = [&](int buf, int kc) {
#pragma unroll
        for (int qq = 0; qq < 2; ++qq) {
            int c = (qq * 128 + t) * 8;                 // element in 2048-elem chunk
            async_b128(kbase + (size_t)kc * DD + c, &kbuf[buf][c]);
        }
#pragma unroll
        for (int qq = 0; qq < 2; ++qq) {
            int c  = (qq * 128 + t) * 8;
            int d  = c >> 5;
            int kk = c & 31;
            async_b128(vbase + (size_t)d * SS + kc + kk, &vbuf[buf][c]);
        }
    };

    stage(0, 0);                          // prologue: chunk 0 -> buffer 0

    for (int kc = 0; kc < SS; kc += 32) {
        int cur = (kc >> 5) & 1;
        __syncthreads();                  // everyone done READING buf[cur^1]
        if (kc + 32 < SS) {
            stage(cur ^ 1, kc + 32);      // async DMA next chunk (overlaps compute)
            asm volatile("s_wait_asynccnt 4" ::: "memory"); // cur chunk landed
        } else {
            asm volatile("s_wait_asynccnt 0" ::: "memory");
        }
        __syncthreads();                  // all waves' portions of buf[cur] visible

        const __bf16* kc_lds = &kbuf[cur][0];   // [32 keys][64] row-major
        const __bf16* vc_lds = &vbuf[cur][0];   // [64 d][32 keys] row-major

        // ---- S = Q @ K^T for 32 keys (two 16-col fragments) ----
        v16bf kb0lo = load_frag(kc_lds, 0,  DD, 0,  lane);
        v16bf kb0hi = load_frag(kc_lds, 0,  DD, 32, lane);
        v16bf kb1lo = load_frag(kc_lds, 16, DD, 0,  lane);
        v16bf kb1hi = load_frag(kc_lds, 16, DD, 32, lane);
        v8f s0 = {}, s1 = {};
        s0 = __builtin_amdgcn_wmma_f32_16x16x32_bf16(false, qa0, false, kb0lo, (short)0, s0, false, false);
        s0 = __builtin_amdgcn_wmma_f32_16x16x32_bf16(false, qa1, false, kb0hi, (short)0, s0, false, false);
        s1 = __builtin_amdgcn_wmma_f32_16x16x32_bf16(false, qa0, false, kb1lo, (short)0, s1, false, false);
        s1 = __builtin_amdgcn_wmma_f32_16x16x32_bf16(false, qa1, false, kb1hi, (short)0, s1, false, false);

        // ---- online softmax (per query row; DPP reduce across 16-lane group) ----
        float tmax[8];
#pragma unroll
        for (int i = 0; i < 8; ++i) {
            float a = s0[i] * scale, b = s1[i] * scale;
            s0[i] = a; s1[i] = b;
            tmax[i] = rowmax16(fmaxf(a, b));
        }
        float c[8], rsum[8];
#pragma unroll
        for (int i = 0; i < 8; ++i) {
            float mn = fmaxf(m[i], tmax[i]);
            c[i] = __expf(m[i] - mn);
            m[i] = mn;
            float p0 = __expf(s0[i] - mn);
            float p1 = __expf(s1[i] - mn);
            s0[i] = p0; s1[i] = p1;
            rsum[i] = rowsum16(p0 + p1);
        }
#pragma unroll
        for (int i = 0; i < 8; ++i) l[i] = l[i] * c[i] + rsum[i];
#pragma unroll
        for (int j = 0; j < 4; ++j)
#pragma unroll
            for (int i = 0; i < 8; ++i) o[j][i] *= c[i];

        // ---- C-layout P (16x32, f32) -> bf16 A-frag via per-wave LDS ----
#pragma unroll
        for (int i = 0; i < 8; ++i) {
            pw[(rowh + i) * 32 + col]      = (__bf16)s0[i];
            pw[(rowh + i) * 32 + col + 16] = (__bf16)s1[i];
        }
        asm volatile("s_wait_dscnt 0" ::: "memory");
        v16bf pa;
        {
            int half = (lane >> 4) & 1;
            const bf16x8* pl0 = reinterpret_cast<const bf16x8*>(&pw[(lane & 15) * 32 + half * 8]);
            const bf16x8* pl1 = reinterpret_cast<const bf16x8*>(&pw[(lane & 15) * 32 + 16 + half * 8]);
            bf16x8 lo = *pl0, hi = *pl1;
#pragma unroll
            for (int i = 0; i < 8; ++i) { pa[i] = lo[i]; pa[8 + i] = hi[i]; }
        }

        // ---- O += P @ V  (V^T chunk in LDS, row-major [d][32 keys]) ----
#pragma unroll
        for (int j = 0; j < 4; ++j) {
            v16bf vb = load_frag(vc_lds, j * 16, 32, 0, lane);
            o[j] = __builtin_amdgcn_wmma_f32_16x16x32_bf16(
                false, pa, false, vb, (short)0, o[j], false, false);
        }
        asm volatile("s_wait_dscnt 0" ::: "memory");
    }

    // ---- epilogue: normalize, store o into [B*S, E] bf16 at col h*64 ----
    int b = bh >> 4, h = bh & 15;
#pragma unroll
    for (int j = 0; j < 4; ++j)
#pragma unroll
        for (int i = 0; i < 8; ++i) {
            float val = o[j][i] / l[i];
            ob[((size_t)(b * SS + q0 + rowh + i)) * EE + h * DD + j * 16 + col] =
                (__bf16)val;
        }
}

// ---------------------------------------------------------------------------
// Host launch
// ---------------------------------------------------------------------------
extern "C" void kernel_launch(void* const* d_in, const int* in_sizes, int n_in,
                              void* d_out, int out_size, void* d_ws, size_t ws_size,
                              hipStream_t stream) {
    const float* x     = (const float*)d_in[0];   // [B,S,E]
    const float* w_qkv = (const float*)d_in[1];   // [E, 3E]
    const float* w_out = (const float*)d_in[2];   // [E, E]
    const float* b_out = (const float*)d_in[3];   // [E]
    float* out = (float*)d_out;

    char* ws = (char*)d_ws;
    size_t off = 0;
    __bf16* x_bf = (__bf16*)(ws + off); off += (size_t)MROWS * EE * 2;       // 16 MB
    __bf16* wq_t = (__bf16*)(ws + off); off += (size_t)N_QKV * EE * 2;       //  6 MB
    __bf16* wo_t = (__bf16*)(ws + off); off += (size_t)EE * EE * 2;          //  2 MB
    __bf16* q_bf = (__bf16*)(ws + off); off += (size_t)BB * HH * SS * DD * 2;// 16 MB
    __bf16* k_bf = (__bf16*)(ws + off); off += (size_t)BB * HH * SS * DD * 2;// 16 MB
    __bf16* v_t  = (__bf16*)(ws + off); off += (size_t)BB * HH * DD * SS * 2;// 16 MB
    __bf16* o_bf = (__bf16*)(ws + off); off += (size_t)MROWS * EE * 2;       // 16 MB

    // 1) convert x to bf16
    {
        int n = MROWS * EE;
        cvt_f32_bf16<<<n / (256 * 4), 256, 0, stream>>>(x, x_bf, n);
    }
    // 2) transpose-convert weights
    {
        int n = EE * N_QKV;
        transpose_cvt<<<(n + 255) / 256, 256, 0, stream>>>(w_qkv, wq_t, EE, N_QKV);
    }
    {
        int n = EE * EE;
        transpose_cvt<<<(n + 255) / 256, 256, 0, stream>>>(w_out, wo_t, EE, EE);
    }
    // 3) QKV projection GEMM (scatter epilogue)
    {
        dim3 grid(MROWS / 16, N_QKV / (64 * 8));  // (512, 6)
        gemm_wave<0><<<grid, 256, 0, stream>>>(x_bf, wq_t, N_QKV, EE,
                                               q_bf, k_bf, v_t, nullptr, nullptr);
    }
    // 4) flash attention: 4 waves per block, async double-buffered K/V in LDS
    {
        attn_kernel<<<BB * HH * (SS / 64), 128, 0, stream>>>(q_bf, k_bf, v_t, o_bf);
    }
    // 5) output projection GEMM (+bias, f32 out)
    {
        dim3 grid(MROWS / 16, EE / (64 * 8));     // (512, 2)
        gemm_wave<1><<<grid, 256, 0, stream>>>(o_bf, wo_t, EE, EE,
                                               nullptr, nullptr, nullptr, b_out, out);
    }
}
